// Cifar10ConvBNN_76424648065027
// MI455X (gfx1250) — 8-bit matrix operand order repaired and checked by reading, compiled, not run on hardware
//
#include <hip/hip_runtime.h>
#include <cstdint>

typedef int v8i  __attribute__((ext_vector_type(8)));
typedef int v2i_t __attribute__((ext_vector_type(2)));
typedef int v4i_t __attribute__((ext_vector_type(4)));

__device__ __forceinline__ int clampi(int v, int lo, int hi) {
    return v < lo ? lo : (v > hi ? hi : v);
}
__device__ __forceinline__ int8_t sgn8(float v) {
    return v > 0.f ? (int8_t)1 : (v < 0.f ? (int8_t)-1 : (int8_t)0);
}

__device__ __forceinline__ v8i load_bfrag(const int8_t* q) {
    v8i b;
#pragma unroll
    for (int j = 0; j < 2; ++j) {
        v4i_t tv = *(const v4i_t*)(q + 32 * j);
        b[4 * j] = tv[0]; b[4 * j + 1] = tv[1];
        b[4 * j + 2] = tv[2]; b[4 * j + 3] = tv[3];
    }
    return b;
}
__device__ __forceinline__ v8i load_afrag(const int8_t* q) {
    v8i a;
#pragma unroll
    for (int j = 0; j < 2; ++j) {
        v4i_t tv = *(const v4i_t*)(q + 32 * j);
        a[4 * j] = tv[0]; a[4 * j + 1] = tv[1];
        a[4 * j + 2] = tv[2]; a[4 * j + 3] = tv[3];
    }
    return a;
}


__global__ void __launch_bounds__(256)
bin_w_conv_k(const float* __restrict__ w, int8_t* __restrict__ w8, int Co, int Ci) {
    long total = (long)Co * Ci * 9;
    long stride = (long)gridDim.x * blockDim.x;
    for (long i = (long)blockIdx.x * blockDim.x + threadIdx.x; i < total; i += stride) {
        int kk = (int)(i % 9);
        long t = i / 9;
        int ci = (int)(t % Ci);
        int co = (int)(t / Ci);
        w8[(long)co * Ci * 9 + (long)kk * Ci + ci] = sgn8(w[i]);
    }
}

__global__ void __launch_bounds__(256)
bin_fw1_perm_k(const float* __restrict__ fw, int8_t* __restrict__ w8) {
    long total = 1024L * 8192;
    long stride = (long)gridDim.x * blockDim.x;
    for (long i = (long)blockIdx.x * blockDim.x + threadIdx.x; i < total; i += stride) {
        int k = (int)(i % 8192);
        int o = (int)(i / 8192);
        int c = k >> 4;
        int hw = k & 15;
        w8[(long)o * 8192 + (long)hw * 512 + c] = sgn8(fw[i]);
    }
}

__global__ void __launch_bounds__(256)
bin_fc_k(const float* __restrict__ fw, int8_t* __restrict__ w8, long total) {
    long stride = (long)gridDim.x * blockDim.x;
    for (long i = (long)blockIdx.x * blockDim.x + threadIdx.x; i < total; i += stride)
        w8[i] = sgn8(fw[i]);
}

__global__ void __launch_bounds__(256)
bin_fw3_pad_k(const float* __restrict__ fw, int8_t* __restrict__ w8) {
    long total = 16L * 1024;
    long stride = (long)gridDim.x * blockDim.x;
    for (long i = (long)blockIdx.x * blockDim.x + threadIdx.x; i < total; i += stride) {
        int k = (int)(i % 1024);
        int o = (int)(i / 1024);
        w8[i] = (o < 10) ? sgn8(fw[(long)o * 1024 + k]) : (int8_t)0;
    }
}

__global__ void __launch_bounds__(256)
conv1_f32_k(const float* __restrict__ x, const float* __restrict__ w,
            const float* __restrict__ bias, float* __restrict__ out,
            int Nn, int H, int W, int Ci, int Co) {
    long total = (long)Nn * H * W * Co;
    long stride = (long)gridDim.x * blockDim.x;
    for (long i = (long)blockIdx.x * blockDim.x + threadIdx.x; i < total; i += stride) {
        int co = (int)(i % Co);
        long t = i / Co;
        int xw = (int)(t % W); t /= W;
        int yh = (int)(t % H);
        int n  = (int)(t / H);
        float s = bias[co];
        for (int ci = 0; ci < Ci; ++ci) {
            for (int kh = 0; kh < 3; ++kh) {
                int yy = clampi(yh + kh - 1, 0, H - 1);
                for (int kw = 0; kw < 3; ++kw) {
                    int xx = clampi(xw + kw - 1, 0, W - 1);
                    float wv = w[((co * Ci + ci) * 3 + kh) * 3 + kw];
                    float sg = (wv > 0.f) ? 1.f : ((wv < 0.f) ? -1.f : 0.f);
                    s += sg * x[((long)(n * Ci + ci) * H + yy) * W + xx];
                }
            }
        }
        out[i] = s;
    }
}

__global__ void __launch_bounds__(256)
conv_bin_wmma_k(const int8_t* __restrict__ act, const int8_t* __restrict__ w8,
                const float* __restrict__ bias, float* __restrict__ out,
                int Nn, int H, int W, int Ci, int Co) {
    const int K = Ci * 9;
    const int P = Nn * H * W;
    const int mtiles = P >> 4;
    const int nquads = Co >> 6;
    const long total = (long)mtiles * nquads;

    long wave = (long)blockIdx.x * (blockDim.x >> 5) + (threadIdx.x >> 5);
    if (wave >= total) return;
    int lane = threadIdx.x & 31;
    int mt = (int)(wave / nquads);
    int nq = (int)(wave - (long)mt * nquads);

    int mrow = lane & 15;
    int half = lane >> 4;
    int aoff = half * 16;
    int boff = half * 16;

    int p = mt * 16 + mrow;
    int n = p / (H * W);
    int rem = p - n * (H * W);
    int y = rem / W;
    int x = rem - y * W;

    const int8_t* bbase = w8 + (size_t)(nq * 64 + mrow) * K + boff;
    const size_t bsub = (size_t)16 * K;

    v8i acc[4];
#pragma unroll
    for (int s = 0; s < 4; ++s) acc[s] = (v8i){0, 0, 0, 0, 0, 0, 0, 0};

    const int cblocks = Ci >> 6;
#pragma unroll
    for (int seg = 0; seg < 9; ++seg) {
        const int kh = seg / 3, kw = seg % 3;
        int yy = clampi(y + kh - 1, 0, H - 1);
        int xx = clampi(x + kw - 1, 0, W - 1);
        const int8_t* ap = act + ((size_t)(n * H + yy) * W + xx) * Ci + aoff;
        const int8_t* bp = bbase + (size_t)seg * Ci;
        for (int cb = 0; cb < cblocks; ++cb) {
            v8i a = load_afrag(ap + (size_t)cb * 64);
            const int8_t* bq = bp + (size_t)cb * 64;
            v8i b0 = load_bfrag(bq);
            v8i b1 = load_bfrag(bq + bsub);
            v8i b2 = load_bfrag(bq + 2 * bsub);
            v8i b3 = load_bfrag(bq + 3 * bsub);
            acc[0] = __builtin_amdgcn_wmma_i32_16x16x64_iu8(true, a, true, b0,
                                                            acc[0], false, false);
            acc[1] = __builtin_amdgcn_wmma_i32_16x16x64_iu8(true, a, true, b1,
                                                            acc[1], false, false);
            acc[2] = __builtin_amdgcn_wmma_i32_16x16x64_iu8(true, a, true, b2,
                                                            acc[2], false, false);
            acc[3] = __builtin_amdgcn_wmma_i32_16x16x64_iu8(true, a, true, b3,
                                                            acc[3], false, false);
        }
    }

    long prow = (long)mt * 16 + half * 8;
#pragma unroll
    for (int s = 0; s < 4; ++s) {
        int ncol = nq * 64 + s * 16 + mrow;
        float bb = bias[ncol];
#pragma unroll
        for (int r = 0; r < 8; ++r)
            out[(prow + r) * Co + ncol] = (float)acc[s][r] + bb;
    }
}

template <int NT>
__global__ void __launch_bounds__(256)
fc_bin_wmma_k(const int8_t* __restrict__ A8, const int8_t* __restrict__ B8,
              const float* __restrict__ bias, float* __restrict__ out,
              int M, int Nf, int K, int Nvalid) {
    const int mtiles = M >> 4;
    const int ngroups = Nf / (16 * NT);
    const long total = (long)mtiles * ngroups;

    long wave = (long)blockIdx.x * (blockDim.x >> 5) + (threadIdx.x >> 5);
    if (wave >= total) return;
    int lane = threadIdx.x & 31;
    int mt = (int)(wave / ngroups);
    int ng = (int)(wave - (long)mt * ngroups);

    int mrow = lane & 15;
    int half = lane >> 4;

    const int8_t* arow = A8 + (size_t)(mt * 16 + mrow) * K + half * 16;
    const int8_t* bbase = B8 + (size_t)(ng * 16 * NT + mrow) * K + half * 16;
    const size_t bsub = (size_t)16 * K;

    v8i acc[NT];
#pragma unroll
    for (int s = 0; s < NT; ++s) acc[s] = (v8i){0, 0, 0, 0, 0, 0, 0, 0};

    const int kblocks = K >> 6;
    for (int kb = 0; kb < kblocks; ++kb) {
        size_t k0 = (size_t)kb << 6;
        v8i a = load_afrag(arow + k0);
#pragma unroll
        for (int s = 0; s < NT; ++s) {
            v8i b = load_bfrag(bbase + s * bsub + k0);
            acc[s] = __builtin_amdgcn_wmma_i32_16x16x64_iu8(true, a, true, b,
                                                            acc[s], false, false);
        }
    }

    long prow = (long)mt * 16 + half * 8;
#pragma unroll
    for (int s = 0; s < NT; ++s) {
        int ncol = ng * 16 * NT + s * 16 + mrow;
        if (ncol < Nvalid) {
            float bb = bias[ncol];
#pragma unroll
            for (int r = 0; r < 8; ++r)
                out[(prow + r) * Nvalid + ncol] = (float)acc[s][r] + bb;
        }
    }
}

__global__ void __launch_bounds__(256)
zero_f32_k(float* __restrict__ p, int n) {
    int i = blockIdx.x * blockDim.x + threadIdx.x;
    if (i < n) p[i] = 0.f;
}

__global__ void __launch_bounds__(256)
stats_k(const float* __restrict__ f, float* __restrict__ sums,
        float* __restrict__ sumsq, long P, int C) {
    long total = P * (long)C;
    long stride = (long)gridDim.x * blockDim.x;
    long i0 = (long)blockIdx.x * blockDim.x + threadIdx.x;
    if (i0 >= total) return;
    float s = 0.f, q = 0.f;
    for (long i = i0; i < total; i += stride) {
        float v = f[i];
        s += v; q += v * v;
    }
    int c = (int)(i0 % C);
    atomicAdd(&sums[c], s);
    atomicAdd(&sumsq[c], q);
}

__global__ void __launch_bounds__(256)
bn_finalize_k(const float* __restrict__ sums, const float* __restrict__ sumsq,
              const float* __restrict__ g, const float* __restrict__ be,
              float* __restrict__ aC, float* __restrict__ cC,
              int C, float invcount) {
    int c = blockIdx.x * blockDim.x + threadIdx.x;
    if (c >= C) return;
    float mu  = sums[c] * invcount;
    float var = sumsq[c] * invcount - mu * mu;
    float s   = g[c] * rsqrtf(var + 1e-5f);
    aC[c] = s;
    cC[c] = be[c] - mu * s;
}

__global__ void __launch_bounds__(256)
bin_act_k(const float* __restrict__ f, const float* __restrict__ aC,
          const float* __restrict__ cC, int8_t* __restrict__ o,
          long P, int C) {
    long total = P * (long)C;
    long stride = (long)gridDim.x * blockDim.x;
    for (long i = (long)blockIdx.x * blockDim.x + threadIdx.x; i < total; i += stride) {
        int c = (int)(i % C);
        o[i] = sgn8(f[i] * aC[c] + cC[c]);
    }
}

__global__ void __launch_bounds__(256)
bn_apply_k(const float* __restrict__ f, const float* __restrict__ aC,
           const float* __restrict__ cC, float* __restrict__ o,
           long P, int C) {
    long total = P * (long)C;
    long stride = (long)gridDim.x * blockDim.x;
    for (long i = (long)blockIdx.x * blockDim.x + threadIdx.x; i < total; i += stride) {
        int c = (int)(i % C);
        o[i] = f[i] * aC[c] + cC[c];
    }
}

__global__ void __launch_bounds__(256)
pool_k(const float* __restrict__ in, float* __restrict__ out,
       int Nn, int H, int W, int C) {
    int Ho = H >> 1, Wo = W >> 1;
    long total = (long)Nn * Ho * Wo * C;
    long stride = (long)gridDim.x * blockDim.x;
    for (long i = (long)blockIdx.x * blockDim.x + threadIdx.x; i < total; i += stride) {
        int c = (int)(i % C);
        long t = i / C;
        int xo = (int)(t % Wo); t /= Wo;
        int yo = (int)(t % Ho);
        int n  = (int)(t / Ho);
        const float* p = in + (((long)(n * H + yo * 2) * W + xo * 2) * C + c);
        long rs = (long)W * C;
        float v0 = p[0], v1 = p[C], v2 = p[rs], v3 = p[rs + C];
        out[i] = fmaxf(fmaxf(v0, v1), fmaxf(v2, v3));
    }
}

static inline int eb(long total) {
    long b = (total + 255) / 256;
    return (int)(b > 4096 ? 4096 : (b < 1 ? 1 : b));
}

extern "C" void kernel_launch(void* const* d_in, const int* in_sizes, int n_in,
                              void* d_out, int out_size, void* d_ws, size_t ws_size,
                              hipStream_t stream) {
    (void)in_sizes; (void)n_in; (void)out_size; (void)ws_size;

    const float* x = (const float*)d_in[0];
    const float *w[7], *b[7], *g[7], *be[7];
    for (int l = 1; l <= 6; ++l) {
        int base = 1 + (l - 1) * 4;
        w[l]  = (const float*)d_in[base + 0];
        b[l]  = (const float*)d_in[base + 1];
        g[l]  = (const float*)d_in[base + 2];
        be[l] = (const float*)d_in[base + 3];
    }
    const float *fw[4], *fb[4], *fg[4], *fbe[4];
    for (int l = 1; l <= 3; ++l) {
        int base = 25 + (l - 1) * 4;
        fw[l]  = (const float*)d_in[base + 0];
        fb[l]  = (const float*)d_in[base + 1];
        fg[l]  = (const float*)d_in[base + 2];
        fbe[l] = (const float*)d_in[base + 3];
    }

    char* ws = (char*)d_ws;
    size_t off = 0;
    auto alloc = [&](size_t bytes) -> size_t {
        size_t r = off;
        off += (bytes + 255) & ~(size_t)255;
        return r;
    };
    float*  convf = (float*)(ws + alloc(256L * 32 * 32 * 128 * 4));
    float*  poolf = (float*)(ws + alloc(256L * 16 * 16 * 128 * 4));
    int8_t* actA  = (int8_t*)(ws + alloc(256L * 32 * 32 * 128));
    int8_t* actB  = (int8_t*)(ws + alloc(256L * 32 * 32 * 128));
    int8_t* w8c[7];
    const int convCi[7] = {0, 3, 128, 128, 256, 256, 512};
    const int convCo[7] = {0, 128, 128, 256, 256, 512, 512};
    for (int l = 2; l <= 6; ++l)
        w8c[l] = (int8_t*)(ws + alloc((size_t)convCo[l] * convCi[l] * 9));
    int8_t* w8f1 = (int8_t*)(ws + alloc(1024L * 8192));
    int8_t* w8f2 = (int8_t*)(ws + alloc(1024L * 1024));
    int8_t* w8f3 = (int8_t*)(ws + alloc(16L * 1024));
    float* fcf   = (float*)(ws + alloc(256L * 1024 * 4));
    float* fc3f  = (float*)(ws + alloc(256L * 16 * 4));
    float* sums  = (float*)(ws + alloc(2048 * 4));
    float* sumsq = sums + 1024;
    float* aC    = (float*)(ws + alloc(1024 * 4));
    float* cC    = (float*)(ws + alloc(1024 * 4));

    for (int l = 2; l <= 6; ++l)
        bin_w_conv_k<<<eb((long)convCo[l] * convCi[l] * 9), 256, 0, stream>>>(
            w[l], w8c[l], convCo[l], convCi[l]);
    bin_fw1_perm_k<<<4096, 256, 0, stream>>>(fw[1], w8f1);
    bin_fc_k<<<4096, 256, 0, stream>>>(fw[2], w8f2, 1024L * 1024);
    bin_fw3_pad_k<<<eb(16L * 1024), 256, 0, stream>>>(fw[3], w8f3);

    auto bn_sign = [&](const float* src, long P, int C,
                       const float* gg, const float* bb, int8_t* dst) {
        zero_f32_k<<<8, 256, 0, stream>>>(sums, 2048);
        stats_k<<<C, 256, 0, stream>>>(src, sums, sumsq, P, C);
        bn_finalize_k<<<(C + 255) / 256, 256, 0, stream>>>(
            sums, sumsq, gg, bb, aC, cC, C, 1.f / (float)P);
        bin_act_k<<<eb(P * C), 256, 0, stream>>>(src, aC, cC, dst, P, C);
    };
    auto conv_bin = [&](const int8_t* act, int layer, float* outf, int H, int W) {
        long waves = ((long)256 * H * W / 16) * (convCo[layer] / 64);
        int blocks = (int)((waves + 7) / 8);
        conv_bin_wmma_k<<<blocks, 256, 0, stream>>>(
            act, w8c[layer], b[layer], outf, 256, H, W, convCi[layer], convCo[layer]);
    };

    conv1_f32_k<<<4096, 256, 0, stream>>>(x, w[1], b[1], convf, 256, 32, 32, 3, 128);
    bn_sign(convf, 256L * 32 * 32, 128, g[1], be[1], actA);

    conv_bin(actA, 2, convf, 32, 32);
    pool_k<<<4096, 256, 0, stream>>>(convf, poolf, 256, 32, 32, 128);
    bn_sign(poolf, 256L * 16 * 16, 128, g[2], be[2], actB);

    conv_bin(actB, 3, convf, 16, 16);
    bn_sign(convf, 256L * 16 * 16, 256, g[3], be[3], actA);

    conv_bin(actA, 4, convf, 16, 16);
    pool_k<<<4096, 256, 0, stream>>>(convf, poolf, 256, 16, 16, 256);
    bn_sign(poolf, 256L * 8 * 8, 256, g[4], be[4], actB);

    conv_bin(actB, 5, convf, 8, 8);
    bn_sign(convf, 256L * 8 * 8, 512, g[5], be[5], actA);

    conv_bin(actA, 6, convf, 8, 8);
    pool_k<<<4096, 256, 0, stream>>>(convf, poolf, 256, 8, 8, 512);
    bn_sign(poolf, 256L * 4 * 4, 512, g[6], be[6], actB);

    {
        long waves = (256L / 16) * (1024 / 64);
        fc_bin_wmma_k<4><<<(int)((waves + 7) / 8), 256, 0, stream>>>(
            actB, w8f1, fb[1], fcf, 256, 1024, 8192, 1024);
    }
    bn_sign(fcf, 256, 1024, fg[1], fbe[1], actA);

    {
        long waves = (256L / 16) * (1024 / 64);
        fc_bin_wmma_k<4><<<(int)((waves + 7) / 8), 256, 0, stream>>>(
            actA, w8f2, fb[2], fcf, 256, 1024, 1024, 1024);
    }
    bn_sign(fcf, 256, 1024, fg[2], fbe[2], actB);

    {
        long waves = (256L / 16) * (16 / 16);
        fc_bin_wmma_k<1><<<(int)((waves + 7) / 8), 256, 0, stream>>>(
            actB, w8f3, fb[3], fc3f, 256, 16, 1024, 10);
    }
    zero_f32_k<<<8, 256, 0, stream>>>(sums, 2048);
    stats_k<<<10, 256, 0, stream>>>(fc3f, sums, sumsq, 256, 10);
    bn_finalize_k<<<1, 256, 0, stream>>>(sums, sumsq, fg[3], fbe[3], aC, cC, 10,
                                         1.f / 256.f);
    bn_apply_k<<<eb(2560), 256, 0, stream>>>(fc3f, aC, cC, (float*)d_out, 256, 10);
}
